// InvertedResidual_85289460564068
// MI455X (gfx1250) — compile-verified, matrix operand order checked by reading, not run on hardware
//
#include <hip/hip_runtime.h>
#include <hip/hip_bf16.h>

// ---------------- problem constants ----------------
constexpr int B_   = 16;
constexpr int INP  = 32;
constexpr int OUP  = 64;
constexpr int HID  = 192;
constexpr int H_   = 96;
constexpr int W_   = 96;
constexpr int HW   = H_ * W_;        // 9216
constexpr int TC   = 48;             // target cols
constexpr int TR   = 48;             // target rows
constexpr int OHW  = TR * TC;        // 2304

// ---------------- workspace layout (float elements) ----------------
constexpr size_t Y_OFF    = 0;
constexpr size_t Y_ELEMS  = (size_t)B_ * HID * HW;            // 28,311,552
constexpr size_t FF_OFF   = Y_OFF;                            // reuse y region
constexpr size_t Z1_OFF   = FF_OFF + (size_t)B_ * HID * OHW;  // reuse y region
constexpr size_t MF_OFF   = Y_OFF + Y_ELEMS;
constexpr size_t MF_ELEMS = (size_t)B_ * HID * H_ * TC;       // 14,155,776
constexpr size_t YP_OFF   = MF_OFF + MF_ELEMS;
constexpr size_t YP_ELEMS = (size_t)B_ * HID * 36;
constexpr size_t SE6_OFF  = YP_OFF + YP_ELEMS;
constexpr size_t SE6_ELEMS= (size_t)B_ * 36;
constexpr size_t SEUP_OFF = SE6_OFF + SE6_ELEMS;
constexpr size_t SEUP_ELEMS=(size_t)B_ * HW;
constexpr size_t IDX_OFF  = SEUP_OFF + SEUP_ELEMS;            // int buffer
constexpr size_t IDX_ELEMS= (size_t)B_ * H_ * TC;
constexpr size_t T_OFF    = IDX_OFF + IDX_ELEMS;
constexpr size_t MM_OFF   = T_OFF + IDX_ELEMS;
constexpr size_t I2_OFF   = MM_OFF + IDX_ELEMS;               // int buffer
constexpr size_t I2_ELEMS = (size_t)B_ * TC * TR;
constexpr size_t T2_OFF   = I2_OFF + I2_ELEMS;
// pre-converted bf16 weights + folded BN (scale, shift) tables
constexpr size_t W1BF_OFF = T2_OFF + I2_ELEMS;                // 6144 bf16 = 3072 f32 slots
constexpr size_t W2BF_OFF = W1BF_OFF + 3072;                  // 12288 bf16 = 6144 f32 slots
constexpr size_t SC1_OFF  = W2BF_OFF + 6144;
constexpr size_t SH1_OFF  = SC1_OFF + HID;
constexpr size_t SC2_OFF  = SH1_OFF + HID;
constexpr size_t SH2_OFF  = SC2_OFF + HID;
constexpr size_t SC3_OFF  = SH2_OFF + HID;
constexpr size_t SH3_OFF  = SC3_OFF + OUP;

// ---------------- vector types ----------------
typedef __attribute__((ext_vector_type(16))) __bf16    v16bf;
typedef __attribute__((ext_vector_type(8)))  float     v8f;
typedef __attribute__((ext_vector_type(4)))  unsigned  v4u;
typedef __attribute__((ext_vector_type(8)))  unsigned  v8u;

__device__ __forceinline__ __bf16 f2bf(float f) {
  unsigned u = __builtin_bit_cast(unsigned, f);
  u += 0x7FFFu + ((u >> 16) & 1u);                 // round-to-nearest-even
  unsigned short h = (unsigned short)(u >> 16);
  return __builtin_bit_cast(__bf16, h);
}

// K packing for the CDNA5 bf16 16x16x32 A/B fragment:
// lanes 0-15 hold K {0..7,16..23}, lanes 16-31 hold K {8..15,24..31}
__device__ __forceinline__ int kbase(int v, int half) {
  return ((v < 4) ? (2 * v) : (2 * v + 8)) + 8 * half;
}

// ---- Tensor Data Mover: DMA a 1-row tile of `nelem` 2-byte elements from
// global `gptr` into LDS offset `lds_off`. D# packed per CDNA5 ISA 8.3/8.4.
__device__ __forceinline__ void tdm_load_row_to_lds(const void* gptr,
                                                    unsigned lds_off,
                                                    unsigned nelem) {
  const unsigned long long ga = (unsigned long long)gptr;
  v4u g0;
  g0[0] = 1u;                                        // count=1 (valid user D#)
  g0[1] = lds_off;                                   // LDS byte address
  g0[2] = (unsigned)(ga & 0xffffffffu);              // global_addr[31:0]
  g0[3] = (unsigned)((ga >> 32) & 0x1ffffffu) | (2u << 30); // addr[56:32] | type=2
  v8u g1;
  g1[0] = 1u << 16;                                  // data_size=1 (2 bytes)
  g1[1] = (nelem & 0xffffu) << 16;                   // tensor_dim0[15:0]
  g1[2] = (nelem >> 16) | (1u << 16);                // tensor_dim0[31:16] | tensor_dim1=1
  g1[3] = (nelem & 0xffffu) << 16;                   // tile_dim0 (row of nelem)
  g1[4] = 0u;                                        // tile_dim1=0, tile_dim2=0
  g1[5] = nelem;                                     // tensor_dim0_stride[31:0]
  g1[6] = 0u;
  g1[7] = 0u;
  v4u gz = {0u, 0u, 0u, 0u};
  asm volatile("tensor_load_to_lds %0, %1, %2, %3"
               :: "s"(g0), "s"(g1), "s"(gz), "s"(gz)
               : "memory");
}

// ================= K0: prep — bf16 weights + folded BN (scale, shift) ============
__global__ __launch_bounds__(256) void k0_prep(
    const float* __restrict__ w1, const float* __restrict__ w2,
    const float* __restrict__ g1, const float* __restrict__ b1,
    const float* __restrict__ m1, const float* __restrict__ v1,
    const float* __restrict__ g2, const float* __restrict__ b2,
    const float* __restrict__ m2, const float* __restrict__ v2,
    const float* __restrict__ g3, const float* __restrict__ b3,
    const float* __restrict__ m3, const float* __restrict__ v3,
    __bf16* __restrict__ w1bf, __bf16* __restrict__ w2bf,
    float* __restrict__ sc1, float* __restrict__ sh1,
    float* __restrict__ sc2, float* __restrict__ sh2,
    float* __restrict__ sc3, float* __restrict__ sh3)
{
  const int t = blockIdx.x * 256 + threadIdx.x;
  const int stride = gridDim.x * 256;
  for (int i = t; i < HID * INP; i += stride) w1bf[i] = f2bf(w1[i]);
  for (int i = t; i < OUP * HID; i += stride) w2bf[i] = f2bf(w2[i]);
  if (t < HID) {
    float s = g1[t] * rsqrtf(v1[t] + 1e-5f);
    sc1[t] = s; sh1[t] = b1[t] - m1[t] * s;
    float s2 = g2[t] * rsqrtf(v2[t] + 1e-5f);
    sc2[t] = s2; sh2[t] = b2[t] - m2[t] * s2;
  }
  if (t < OUP) {
    float s3 = g3[t] * rsqrtf(v3[t] + 1e-5f);
    sc3[t] = s3; sh3[t] = b3[t] - m3[t] * s3;
  }
}

// ================= K1: 1x1 conv 32->192 + BN + ReLU6 (WMMA bf16, TDM-staged A) ===
__global__ __launch_bounds__(256) void k1_pw1_wmma(
    const float* __restrict__ x, const __bf16* __restrict__ w1bf,
    const float* __restrict__ sc1, const float* __restrict__ sh1,
    float* __restrict__ y)
{
  __shared__ __bf16 w1s[HID * INP];                  // 12 KB
  if (threadIdx.x < 32) {                            // wave 0 drives the TDM
    tdm_load_row_to_lds(w1bf, (unsigned)(unsigned long long)(uintptr_t)w1s,
                        HID * INP);
    __builtin_amdgcn_s_wait_tensorcnt(0);
  }
  __syncthreads();

  const int wave = threadIdx.x >> 5;
  const int lane = threadIdx.x & 31;
  const int half = lane >> 4;
  const int idx  = lane & 15;
  const int ntile = blockIdx.x * 8 + wave;           // one 16-pixel tile / wave
  const int pix0  = ntile * 16;
  const int b     = pix0 / HW;
  const int my_hw = (pix0 % HW) + idx;

  // B fragment: x[b, k, hw] per bf16 32x16 layout
  v16bf bf;
  const float* xb = x + (size_t)b * INP * HW + my_hw;
#pragma unroll
  for (int v = 0; v < 8; ++v) {
    const int kb = kbase(v, half);
    bf[2 * v]     = f2bf(xb[(size_t)kb * HW]);
    bf[2 * v + 1] = f2bf(xb[(size_t)(kb + 1) * HW]);
  }

  for (int mt = 0; mt < HID / 16; ++mt) {            // 12 M-tiles reuse B fragment
    v16bf af;
    const int row = mt * 16 + idx;
#pragma unroll
    for (int v = 0; v < 8; ++v) {
      const int kb = kbase(v, half);
      af[2 * v]     = w1s[row * INP + kb];
      af[2 * v + 1] = w1s[row * INP + kb + 1];
    }
    v8f acc = {};
    acc = __builtin_amdgcn_wmma_f32_16x16x32_bf16(false, af, false, bf,
                                                  (short)0, acc, false, false);
#pragma unroll
    for (int r = 0; r < 8; ++r) {
      const int o = mt * 16 + r + 8 * half;
      float val = fmaf(acc[r], sc1[o], sh1[o]);
      val = fminf(fmaxf(val, 0.f), 6.f);
      y[((size_t)b * HID + o) * HW + my_hw] = val;
    }
  }
}

// ================= K2: 6x6 avg pool (16x16 cells), one wave per cell =============
__global__ __launch_bounds__(256) void k2_pool(const float* __restrict__ y,
                                               float* __restrict__ yp)
{
  const int wave = threadIdx.x >> 5, lane = threadIdx.x & 31;
  const int cell = blockIdx.x * 8 + wave;            // ((b*HID+c)*6+cy)*6+cx
  const int cx = cell % 6, cy = (cell / 6) % 6;
  const int c  = (cell / 36) % HID, b = cell / (36 * HID);
  const float* base = y + ((size_t)b * HID + c) * HW + (cy * 16) * W_ + cx * 16;
  float s = 0.f;
#pragma unroll
  for (int q = 0; q < 8; ++q) {
    const int lin = q * 32 + lane;
    s += base[(lin >> 4) * W_ + (lin & 15)];
  }
  for (int off = 16; off; off >>= 1) s += __shfl_xor(s, off, 32);
  if (lane == 0) yp[cell] = s * (1.f / 256.f);
}

// ================= K3: SE squeeze 192->1 + sigmoid, one wave per (b,cell) ========
__global__ __launch_bounds__(256) void k3_se(const float* __restrict__ yp,
                                             const float* __restrict__ se_w,
                                             const float* __restrict__ se_b,
                                             float* __restrict__ se6)
{
  const int wave = threadIdx.x >> 5, lane = threadIdx.x & 31;
  const int id = blockIdx.x * 8 + wave;              // b*36 + cell
  const int cell = id % 36, b = id / 36;
  float s = 0.f;
  for (int c = lane; c < HID; c += 32)
    s += yp[((size_t)b * HID + c) * 36 + cell] * se_w[c];
  for (int off = 16; off; off >>= 1) s += __shfl_xor(s, off, 32);
  if (lane == 0) se6[id] = 1.f / (1.f + __expf(-(s + se_b[0])));
}

// ================= K4: bilinear (align-corners) upsample 6x6 -> 96x96 ============
__global__ void k4_upsample(const float* __restrict__ se6, float* __restrict__ seup)
{
  const int t = blockIdx.x * blockDim.x + threadIdx.x;
  if (t >= B_ * HW) return;
  const int w = t % W_, h = (t / W_) % H_, b = t / HW;
  const float sc = 5.f / 95.f;
  const float ch = h * sc; int r0 = min((int)floorf(ch), 4); const float fr = ch - r0;
  const float cw = w * sc; int c0 = min((int)floorf(cw), 4); const float fc = cw - c0;
  const float* s = se6 + (size_t)b * 36;
  const float top = s[r0 * 6 + c0] * (1.f - fc) + s[r0 * 6 + c0 + 1] * fc;
  const float bot = s[(r0 + 1) * 6 + c0] * (1.f - fc) + s[(r0 + 1) * 6 + c0 + 1] * fc;
  seup[t] = top * (1.f - fr) + bot * fr;
}

// ================= K5: per-row cumsum + horizontal query indices =================
__global__ void k5_rowprep(const float* __restrict__ seup, int* __restrict__ idxb,
                           float* __restrict__ tb, float* __restrict__ mm)
{
  const int t = blockIdx.x * blockDim.x + threadIdx.x;
  if (t >= B_ * H_) return;
  const int h = t % H_, b = t / H_;
  const float* ys = seup + (size_t)b * HW + h * W_;
  float S = 0.f;
  for (int w = 0; w < W_; ++w) S += ys[w] + 0.001f;
  const float alpha = (W_ * 0.5f) / S;
  float xx[W_];
  float run = 0.f;
  for (int w = 0; w < W_; ++w) { run += alpha * (ys[w] + 0.001f) + 0.5f; xx[w] = run; }
  for (int j = 0; j < TC; ++j) {
    const float cq = 2.f * j;
    int cnt = 0;
    for (int w = 0; w < W_; ++w) cnt += (xx[w] < cq) ? 1 : 0;
    const int i = min(max(cnt - 1, 0), W_ - 2);
    const float tt = (cq - xx[i]) / (xx[i + 1] - xx[i]);
    const size_t o = (size_t)t * TC + j;
    idxb[o] = i;
    tb[o]   = tt;
    mm[o]   = ys[i] + (ys[i + 1] - ys[i]) * tt;
  }
}

// ================= K6: horizontal gather-interp of y*se -> mf ====================
__global__ void k6_mf(const float* __restrict__ y, const float* __restrict__ seup,
                      const int* __restrict__ idxb, const float* __restrict__ tb,
                      float* __restrict__ mf)
{
  const size_t t = (size_t)blockIdx.x * blockDim.x + threadIdx.x;
  const size_t N = (size_t)B_ * HID * H_ * TC;
  if (t >= N) return;
  const int j = (int)(t % TC);
  const int h = (int)((t / TC) % H_);
  const int c = (int)((t / ((size_t)TC * H_)) % HID);
  const int b = (int)(t / ((size_t)TC * H_ * HID));
  const size_t ro = ((size_t)b * H_ + h) * TC + j;
  const int   i  = idxb[ro];
  const float tt = tb[ro];
  const float* yr = y    + ((size_t)b * HID + c) * HW + (size_t)h * W_;
  const float* sr = seup + (size_t)b * HW + (size_t)h * W_;
  const float y0 = yr[i] * sr[i];
  const float y1 = yr[i + 1] * sr[i + 1];
  mf[t] = y0 + (y1 - y0) * tt;
}

// ================= K7: per-column cumsum + vertical query indices ================
__global__ void k7_colprep(const float* __restrict__ mm, int* __restrict__ i2b,
                           float* __restrict__ t2b)
{
  const int t = blockIdx.x * blockDim.x + threadIdx.x;
  if (t >= B_ * TC) return;
  const int j = t % TC, b = t / TC;
  const float* col = mm + (size_t)b * H_ * TC + j;   // stride TC
  float S = 0.f;
  for (int r = 0; r < H_; ++r) S += col[(size_t)r * TC] + 0.001f;
  const float beta = (H_ * 0.5f) / S;
  float yy[H_];
  float run = 0.f;
  for (int r = 0; r < H_; ++r) { run += beta * (col[(size_t)r * TC] + 0.001f) + 0.5f; yy[r] = run; }
  for (int k = 0; k < TR; ++k) {
    const float rq = 2.f * k;
    int cnt = 0;
    for (int r = 0; r < H_; ++r) cnt += (yy[r] < rq) ? 1 : 0;
    const int i = min(max(cnt - 1, 0), H_ - 2);
    const float t2 = (rq - yy[i]) / (yy[i + 1] - yy[i]);
    const size_t o = ((size_t)b * TC + j) * TR + k;
    i2b[o] = i;
    t2b[o] = t2;
  }
}

// ================= K8: vertical gather-interp -> ff ==============================
__global__ void k8_ff(const float* __restrict__ mf, const int* __restrict__ i2b,
                      const float* __restrict__ t2b, float* __restrict__ ff)
{
  const size_t t = (size_t)blockIdx.x * blockDim.x + threadIdx.x;
  const size_t N = (size_t)B_ * HID * TR * TC;
  if (t >= N) return;
  const int j = (int)(t % TC);
  const int k = (int)((t / TC) % TR);
  const int c = (int)((t / (TC * TR)) % HID);
  const int b = (int)(t / ((size_t)TC * TR * HID));
  const size_t o2 = ((size_t)b * TC + j) * TR + k;
  const int   i  = i2b[o2];
  const float tt = t2b[o2];
  const float* mc = mf + ((size_t)b * HID + c) * H_ * TC + j;
  const float f0 = mc[(size_t)i * TC];
  const float f1 = mc[(size_t)(i + 1) * TC];
  ff[t] = f0 + (f1 - f0) * tt;
}

// ================= K9: depthwise 3x3 SAME + BN + ReLU6 ===========================
__global__ void k9_dw(const float* __restrict__ ff, const float* __restrict__ wdw,
                      const float* __restrict__ sc2, const float* __restrict__ sh2,
                      float* __restrict__ z1)
{
  const size_t t = (size_t)blockIdx.x * blockDim.x + threadIdx.x;
  const size_t N = (size_t)B_ * HID * TR * TC;
  if (t >= N) return;
  const int j = (int)(t % TC);
  const int k = (int)((t / TC) % TR);
  const int c = (int)((t / (TC * TR)) % HID);
  const int b = (int)(t / ((size_t)TC * TR * HID));
  const float* img = ff + ((size_t)b * HID + c) * TR * TC;
  const float* wk  = wdw + (size_t)c * 9;
  float s = 0.f;
#pragma unroll
  for (int dy = -1; dy <= 1; ++dy)
#pragma unroll
    for (int dx = -1; dx <= 1; ++dx) {
      const int rr = k + dy, cc = j + dx;
      if (rr >= 0 && rr < TR && cc >= 0 && cc < TC)
        s += img[rr * TC + cc] * wk[(dy + 1) * 3 + (dx + 1)];
    }
  float val = fmaf(s, sc2[c], sh2[c]);
  z1[t] = fminf(fmaxf(val, 0.f), 6.f);
}

// ================= K10: 1x1 conv 192->64 + BN (WMMA bf16, TDM-staged A) ==========
__global__ __launch_bounds__(256) void k10_pw2_wmma(
    const float* __restrict__ z1, const __bf16* __restrict__ w2bf,
    const float* __restrict__ sc3, const float* __restrict__ sh3,
    float* __restrict__ out)
{
  __shared__ __bf16 w2s[OUP * HID];                  // 24 KB
  if (threadIdx.x < 32) {
    tdm_load_row_to_lds(w2bf, (unsigned)(unsigned long long)(uintptr_t)w2s,
                        OUP * HID);
    __builtin_amdgcn_s_wait_tensorcnt(0);
  }
  __syncthreads();

  const int wave = threadIdx.x >> 5, lane = threadIdx.x & 31;
  const int half = lane >> 4, idx = lane & 15;
  const int ntile = blockIdx.x * 8 + wave;
  const int pix0  = ntile * 16;
  const int b     = pix0 / OHW;
  const int p     = (pix0 % OHW) + idx;

  v8f acc[4] = {{}, {}, {}, {}};
  const float* zb = z1 + (size_t)b * HID * OHW + p;

  for (int kc = 0; kc < HID / 32; ++kc) {            // 6 K-chunks
    v16bf bf;
#pragma unroll
    for (int v = 0; v < 8; ++v) {
      const int kb = kc * 32 + kbase(v, half);
      bf[2 * v]     = f2bf(zb[(size_t)kb * OHW]);
      bf[2 * v + 1] = f2bf(zb[(size_t)(kb + 1) * OHW]);
    }
#pragma unroll
    for (int mt = 0; mt < 4; ++mt) {                 // 4 M-tiles (OUP=64)
      v16bf af;
      const int row = mt * 16 + idx;
#pragma unroll
      for (int v = 0; v < 8; ++v) {
        const int kb = kc * 32 + kbase(v, half);
        af[2 * v]     = w2s[row * HID + kb];
        af[2 * v + 1] = w2s[row * HID + kb + 1];
      }
      acc[mt] = __builtin_amdgcn_wmma_f32_16x16x32_bf16(false, af, false, bf,
                                                        (short)0, acc[mt], false, false);
    }
  }
#pragma unroll
  for (int mt = 0; mt < 4; ++mt)
#pragma unroll
    for (int r = 0; r < 8; ++r) {
      const int o = mt * 16 + r + 8 * half;
      out[((size_t)b * OUP + o) * OHW + p] = fmaf(acc[mt][r], sc3[o], sh3[o]);
    }
}

// ================================ launcher =======================================
extern "C" void kernel_launch(void* const* d_in, const int* in_sizes, int n_in,
                              void* d_out, int out_size, void* d_ws, size_t ws_size,
                              hipStream_t stream) {
  const float* x    = (const float*)d_in[0];
  const float* w1   = (const float*)d_in[1];
  const float* g1   = (const float*)d_in[2];
  const float* b1   = (const float*)d_in[3];
  const float* m1   = (const float*)d_in[4];
  const float* v1   = (const float*)d_in[5];
  const float* se_w = (const float*)d_in[6];
  const float* se_b = (const float*)d_in[7];
  const float* wdw  = (const float*)d_in[8];
  const float* g2   = (const float*)d_in[9];
  const float* b2   = (const float*)d_in[10];
  const float* m2   = (const float*)d_in[11];
  const float* v2   = (const float*)d_in[12];
  const float* w2   = (const float*)d_in[13];
  const float* g3   = (const float*)d_in[14];
  const float* b3   = (const float*)d_in[15];
  const float* m3   = (const float*)d_in[16];
  const float* v3   = (const float*)d_in[17];

  float* ws   = (float*)d_ws;
  float* y    = ws + Y_OFF;
  float* ffb  = ws + FF_OFF;   // reuses y region
  float* z1b  = ws + Z1_OFF;   // reuses y region
  float* mf   = ws + MF_OFF;
  float* yp   = ws + YP_OFF;
  float* se6  = ws + SE6_OFF;
  float* seup = ws + SEUP_OFF;
  int*   idxb = (int*)(ws + IDX_OFF);
  float* tbuf = ws + T_OFF;
  float* mmb  = ws + MM_OFF;
  int*   i2b  = (int*)(ws + I2_OFF);
  float* t2b  = ws + T2_OFF;
  __bf16* w1bf = (__bf16*)(ws + W1BF_OFF);
  __bf16* w2bf = (__bf16*)(ws + W2BF_OFF);
  float* sc1 = ws + SC1_OFF; float* sh1 = ws + SH1_OFF;
  float* sc2 = ws + SC2_OFF; float* sh2 = ws + SH2_OFF;
  float* sc3 = ws + SC3_OFF; float* sh3 = ws + SH3_OFF;

  // K0: weight conversion + BN folding
  k0_prep<<<48, 256, 0, stream>>>(w1, w2, g1, b1, m1, v1, g2, b2, m2, v2,
                                  g3, b3, m3, v3, w1bf, w2bf,
                                  sc1, sh1, sc2, sh2, sc3, sh3);
  // K1: 9216 pixel-tiles / 8 waves per block
  k1_pw1_wmma<<<1152, 256, 0, stream>>>(x, w1bf, sc1, sh1, y);
  // K2: B*HID*36 = 110592 cells, 8 waves/block
  k2_pool<<<13824, 256, 0, stream>>>(y, yp);
  // K3: B*36 = 576 waves
  k3_se<<<72, 256, 0, stream>>>(yp, se_w, se_b, se6);
  // K4: B*HW threads
  k4_upsample<<<(B_ * HW + 255) / 256, 256, 0, stream>>>(se6, seup);
  // K5: B*H threads (serial row scans)
  k5_rowprep<<<(B_ * H_ + 255) / 256, 256, 0, stream>>>(seup, idxb, tbuf, mmb);
  // K6: B*HID*H*TC threads
  {
    const size_t n = (size_t)B_ * HID * H_ * TC;
    k6_mf<<<(unsigned)((n + 255) / 256), 256, 0, stream>>>(y, seup, idxb, tbuf, mf);
  }
  // K7: B*TC threads (serial column scans)
  k7_colprep<<<(B_ * TC + 255) / 256, 256, 0, stream>>>(mmb, i2b, t2b);
  // K8: B*HID*TR*TC threads
  {
    const size_t n = (size_t)B_ * HID * TR * TC;
    k8_ff<<<(unsigned)((n + 255) / 256), 256, 0, stream>>>(mf, i2b, t2b, ffb);
  }
  // K9: depthwise conv + BN + ReLU6
  {
    const size_t n = (size_t)B_ * HID * TR * TC;
    k9_dw<<<(unsigned)((n + 255) / 256), 256, 0, stream>>>(ffb, wdw, sc2, sh2, z1b);
  }
  // K10: 2304 pixel-tiles / 8 waves per block
  k10_pw2_wmma<<<288, 256, 0, stream>>>(z1b, w2bf, sc3, sh3, (float*)d_out);
}
